// LSTM_Model_18468359373513
// MI455X (gfx1250) — matrix operand order repaired and checked by reading, not compiled, not run on hardware
//
#include <hip/hip_runtime.h>
#include <stdint.h>

// Problem dims (from reference)
#define BB 64
#define TT 512
#define UU 256    // embedding units
#define HH 512    // hidden
#define FF 64     // final features
#define G4 2048   // 4*H gate width
#define K1 768    // UU + HH  (layer-1 fused K)
#define K2 1024   // HH + HH  (layer-2 fused K)
#define NBLK 128  // persistent grid size (must all be resident)

typedef __attribute__((ext_vector_type(16))) __bf16 v16bf;
typedef __attribute__((ext_vector_type(8)))  __bf16 v8bf;
typedef __attribute__((ext_vector_type(8)))  float  v8f;

__device__ __forceinline__ __bf16 f2bf(float f) {
  union { float f; uint32_t u; } v; v.f = f;
  uint32_t r = v.u + 0x7FFFu + ((v.u >> 16) & 1u);   // round-to-nearest-even
  union { unsigned short s; __bf16 b; } o; o.s = (unsigned short)(r >> 16);
  return o.b;
}

__device__ __forceinline__ float sigmoid_f(float x) { return 1.0f / (1.0f + __expf(-x)); }

__device__ __forceinline__ v8f wmma_bf16(v16bf a, v16bf b, v8f c) {
  // D = A(16x32 bf16) * B(32x16 bf16) + C(16x16 f32)
  return __builtin_amdgcn_wmma_f32_16x16x32_bf16(false, a, false, b, (short)0, c, false, false);
}

// A fragment (16x32 bf16) from row-major LDS tile. Contiguous k order used here (same as B, not ISA table order):
// lane m = lane&15; lanes<16 hold K=0..15, lanes>=16 hold K=16..31.
__device__ __forceinline__ v16bf load_a_lds(const __bf16* row, int kc, int lane) {
  const int kh = (lane >> 4) << 4;                // 0 or 16
  return *(const v16bf*)(row + kc + kh);          // K = kc+kh .. +15
}

// B fragment (32x16 bf16) from N-major (transposed) weights: lane n = lane&15,
// lanes<16 hold K=kc..kc+15, lanes>=16 hold K=kc+16..kc+31 — contiguous 32B.
__device__ __forceinline__ v16bf load_b_nmajor(const __bf16* wcol, int LDK, int kc, int lane) {
  const int n  = lane & 15;
  const int kh = (lane >> 4) << 4;                // 0 or 16
  return *(const v16bf*)(wcol + (size_t)n * LDK + kc + kh);
}

// ---- grid-wide barrier (all NBLK blocks resident) ----
__device__ __forceinline__ void grid_sync(unsigned* cnt, unsigned* gen, unsigned nblk) {
  __syncthreads();
  if (threadIdx.x == 0) {
    __threadfence();
    volatile unsigned* vgen = (volatile unsigned*)gen;
    unsigned g = *vgen;
    if (atomicAdd(cnt, 1u) == nblk - 1u) {
      *((volatile unsigned*)cnt) = 0u;
      __threadfence();
      atomicAdd(gen, 1u);
    } else {
      while (*vgen == g) { __builtin_amdgcn_s_sleep(2); }
      __threadfence();
    }
  }
  __syncthreads();
}

// ===================== setup kernels =====================

__global__ void zero_kernel(uint32_t* p, size_t words) {
  for (size_t i = (size_t)blockIdx.x * blockDim.x + threadIdx.x; i < words;
       i += (size_t)gridDim.x * blockDim.x)
    p[i] = 0u;
}

// dst[n*LD + koff + k] = bf16(src[k*N + n])   (src is K x N row-major f32)
__global__ void pack_transpose_kernel(const float* __restrict__ src, __bf16* __restrict__ dst,
                                      int K, int N, int LD, int koff) {
  const size_t total = (size_t)K * N;
  for (size_t i = (size_t)blockIdx.x * blockDim.x + threadIdx.x; i < total;
       i += (size_t)gridDim.x * blockDim.x) {
    int k = (int)(i / N), n = (int)(i % N);
    dst[(size_t)n * LD + koff + k] = f2bf(src[i]);
  }
}

__global__ void embed_kernel(const int* __restrict__ tokens, const float* __restrict__ emb,
                             __bf16* __restrict__ xb) {
  const size_t total = (size_t)BB * TT * UU;
  for (size_t i = (size_t)blockIdx.x * blockDim.x + threadIdx.x; i < total;
       i += (size_t)gridDim.x * blockDim.x) {
    int u = (int)(i & (UU - 1));
    size_t bt = i >> 8;                 // UU == 256
    int tok = tokens[bt];
    xb[i] = f2bf(emb[(size_t)tok * UU + u]);
  }
}

// ===================== persistent fused 2-layer LSTM =====================
// Grid: 128 WGs x 128 threads (4 waves).  512 waves total -> one 16x16 Z tile each.
// Per step: GEMM1 (K=768) -> sync -> EW1 -> sync -> GEMM2 (K=1024) -> sync -> EW2 -> sync.
__launch_bounds__(128)
__global__ void lstm_seq_kernel(const int* __restrict__ tokens,
                                const float* __restrict__ b1, const float* __restrict__ b2,
                                const __bf16* __restrict__ xb,
                                const __bf16* __restrict__ W1t,   // [2048][768]  bf16
                                const __bf16* __restrict__ W2t,   // [2048][1024] bf16
                                __bf16* __restrict__ h1, float* __restrict__ c1,
                                __bf16* __restrict__ h2, float* __restrict__ c2,
                                float* __restrict__ Z,            // [64][2048] f32
                                unsigned* __restrict__ bar) {
  __shared__ __align__(32) __bf16 smA[16 * K2];   // 32 KB; reused for K1 phase
  const int tid  = threadIdx.x;
  const int lane = tid & 31;
  const int wave = tid >> 5;
  const int tile = blockIdx.x * 4 + wave;   // 0..511
  const int mt = tile >> 7;                 // 0..3   (same for all 4 waves of a WG)
  const int nt = tile & 127;                // 0..127
  const int b0 = mt * 16;
  const __bf16* aRow1 = smA + (size_t)(lane & 15) * K1;
  const __bf16* aRow2 = smA + (size_t)(lane & 15) * K2;
  const __bf16* w1col = W1t + (size_t)(nt * 16) * K1;
  const __bf16* w2col = W2t + (size_t)(nt * 16) * K2;
  unsigned* cnt = bar;
  unsigned* gen = bar + 1;

  for (int t = 0; t < TT; ++t) {
    // ---------------- layer 1: Z = [x_t | h1] @ [W1;U1] ----------------
    for (int i = tid; i < 16 * UU; i += 128) {
      int m = i >> 8, k = i & (UU - 1);
      smA[m * K1 + k] = xb[((size_t)(b0 + m) * TT + t) * UU + k];
    }
    for (int i = tid; i < 16 * HH; i += 128) {
      int m = i >> 9, k = i & (HH - 1);
      smA[m * K1 + UU + k] = h1[(b0 + m) * HH + k];
    }
    __syncthreads();
    {
      v8f acc = {};
      for (int kc = 0; kc < K1; kc += 32) {
        if (kc + 64 < K1)
          __builtin_prefetch(w1col + (size_t)(lane & 15) * K1 + kc + 64, 0, 1);
        v16bf a    = load_a_lds(aRow1, kc, lane);
        v16bf bmat = load_b_nmajor(w1col, K1, kc, lane);
        acc = wmma_bf16(a, bmat, acc);
      }
      const int n  = nt * 16 + (lane & 15);
      const int mh = (lane >> 4) * 8;
#pragma unroll
      for (int v = 0; v < 8; ++v)
        Z[(size_t)(b0 + mh + v) * G4 + n] = acc[v];
    }
    grid_sync(cnt, gen, NBLK);
    // elementwise gates, layer 1
    for (int u = blockIdx.x * 128 + tid; u < BB * HH; u += NBLK * 128) {
      int bi = u >> 9, j = u & (HH - 1);
      const float* zr = Z + (size_t)bi * G4;
      float zi = zr[j]          + b1[j];
      float zf = zr[HH + j]     + b1[HH + j];
      float zg = zr[2 * HH + j] + b1[2 * HH + j];
      float zo = zr[3 * HH + j] + b1[3 * HH + j];
      float ig = sigmoid_f(zi), fg = sigmoid_f(zf), og = sigmoid_f(zo);
      float cn = fg * c1[u] + ig * tanhf(zg);
      float hn = og * tanhf(cn);
      if (tokens[bi * TT + t] != 0) { c1[u] = cn; h1[u] = f2bf(hn); }
    }
    grid_sync(cnt, gen, NBLK);
    // ---------------- layer 2: Z = [h1 | h2] @ [W2;U2] ----------------
    for (int i = tid; i < 16 * HH; i += 128) {
      int m = i >> 9, k = i & (HH - 1);
      smA[m * K2 + k]      = h1[(b0 + m) * HH + k];
      smA[m * K2 + HH + k] = h2[(b0 + m) * HH + k];
    }
    __syncthreads();
    {
      v8f acc = {};
      for (int kc = 0; kc < K2; kc += 32) {
        if (kc + 64 < K2)
          __builtin_prefetch(w2col + (size_t)(lane & 15) * K2 + kc + 64, 0, 1);
        v16bf a    = load_a_lds(aRow2, kc, lane);
        v16bf bmat = load_b_nmajor(w2col, K2, kc, lane);
        acc = wmma_bf16(a, bmat, acc);
      }
      const int n  = nt * 16 + (lane & 15);
      const int mh = (lane >> 4) * 8;
#pragma unroll
      for (int v = 0; v < 8; ++v)
        Z[(size_t)(b0 + mh + v) * G4 + n] = acc[v];
    }
    grid_sync(cnt, gen, NBLK);
    // elementwise gates, layer 2
    for (int u = blockIdx.x * 128 + tid; u < BB * HH; u += NBLK * 128) {
      int bi = u >> 9, j = u & (HH - 1);
      const float* zr = Z + (size_t)bi * G4;
      float zi = zr[j]          + b2[j];
      float zf = zr[HH + j]     + b2[HH + j];
      float zg = zr[2 * HH + j] + b2[2 * HH + j];
      float zo = zr[3 * HH + j] + b2[3 * HH + j];
      float ig = sigmoid_f(zi), fg = sigmoid_f(zf), og = sigmoid_f(zo);
      float cn = fg * c2[u] + ig * tanhf(zg);
      float hn = og * tanhf(cn);
      if (tokens[bi * TT + t] != 0) { c2[u] = cn; h2[u] = f2bf(hn); }
    }
    grid_sync(cnt, gen, NBLK);
  }
}

// ===================== head: out = h2 @ Wd + bd (WMMA) =====================
__launch_bounds__(512)
__global__ void head_kernel(const __bf16* __restrict__ h2,
                            const __bf16* __restrict__ WdT,   // [64][512] bf16 (N-major)
                            const float* __restrict__ bd,
                            float* __restrict__ out) {
  const int tid = threadIdx.x, lane = tid & 31, wave = tid >> 5;  // 16 waves
  const int mt = wave >> 2, nt = wave & 3;                        // 4x4 tiles of 64x64
  const int m  = mt * 16 + (lane & 15);
  const int kb = (lane >> 4) << 3;
  const int kh = (lane >> 4) << 4;
  const __bf16* wcol = WdT + (size_t)(nt * 16 + (lane & 15)) * HH;
  v8f acc = {};
  for (int kc = 0; kc < HH; kc += 32) {
    v16bf a = *(const v16bf*)(h2 + (size_t)m * HH + kc + kh);
    v16bf bm = *(const v16bf*)(wcol + kc + kh);
    acc = wmma_bf16(a, bm, acc);
  }
  const int n  = nt * 16 + (lane & 15);
  const int mh = (lane >> 4) * 8;
#pragma unroll
  for (int v = 0; v < 8; ++v)
    out[(size_t)(mt * 16 + mh + v) * FF + n] = acc[v] + bd[n];
}

// ===================== host launcher =====================
extern "C" void kernel_launch(void* const* d_in, const int* in_sizes, int n_in,
                              void* d_out, int out_size, void* d_ws, size_t ws_size,
                              hipStream_t stream) {
  const int*   tokens = (const int*)d_in[0];
  const float* emb    = (const float*)d_in[1];
  const float* W1     = (const float*)d_in[2];
  const float* U1     = (const float*)d_in[3];
  const float* b1     = (const float*)d_in[4];
  const float* W2     = (const float*)d_in[5];
  const float* U2     = (const float*)d_in[6];
  const float* b2     = (const float*)d_in[7];
  const float* Wd     = (const float*)d_in[8];
  const float* bd     = (const float*)d_in[9];
  float* out = (float*)d_out;

  char* ws = (char*)d_ws;
  size_t off = 0;
  auto take = [&](size_t bytes) { size_t o = off; off += (bytes + 255) & ~(size_t)255; return o; };
  const size_t o_bar = take(256);                                  // barrier cnt/gen
  const size_t o_h1  = take((size_t)BB * HH * 2);                  // bf16
  const size_t o_h2  = take((size_t)BB * HH * 2);                  // bf16
  const size_t o_c1  = take((size_t)BB * HH * 4);                  // f32
  const size_t o_c2  = take((size_t)BB * HH * 4);                  // f32
  const size_t zero_bytes = off;                                   // state to zero each call
  const size_t o_Z   = take((size_t)BB * G4 * 4);                  // f32 gate buffer
  const size_t o_xb  = take((size_t)BB * TT * UU * 2);             // bf16 embedded input
  const size_t o_W1t = take((size_t)G4 * K1 * 2);                  // bf16 [2048][768]
  const size_t o_W2t = take((size_t)G4 * K2 * 2);                  // bf16 [2048][1024]
  const size_t o_Wdt = take((size_t)FF * HH * 2);                  // bf16 [64][512]
  (void)ws_size; (void)in_sizes; (void)n_in; (void)out_size;

  // 1) zero barrier + h/c state (ws is poisoned; must re-init every call)
  zero_kernel<<<256, 256, 0, stream>>>((uint32_t*)ws, zero_bytes / 4);

  // 2) pack weights -> transposed bf16 (K-concatenated for fused GEMMs)
  pack_transpose_kernel<<<512, 256, 0, stream>>>(W1, (__bf16*)(ws + o_W1t), UU, G4, K1, 0);
  pack_transpose_kernel<<<512, 256, 0, stream>>>(U1, (__bf16*)(ws + o_W1t), HH, G4, K1, UU);
  pack_transpose_kernel<<<512, 256, 0, stream>>>(W2, (__bf16*)(ws + o_W2t), HH, G4, K2, 0);
  pack_transpose_kernel<<<512, 256, 0, stream>>>(U2, (__bf16*)(ws + o_W2t), HH, G4, K2, HH);
  pack_transpose_kernel<<<64,  256, 0, stream>>>(Wd, (__bf16*)(ws + o_Wdt), HH, FF, HH, 0);

  // 3) embedding gather -> bf16
  embed_kernel<<<1024, 256, 0, stream>>>(tokens, emb, (__bf16*)(ws + o_xb));

  // 4) persistent fused 2-layer LSTM over 512 timesteps
  lstm_seq_kernel<<<NBLK, 128, 0, stream>>>(
      tokens, b1, b2,
      (const __bf16*)(ws + o_xb),
      (const __bf16*)(ws + o_W1t), (const __bf16*)(ws + o_W2t),
      (__bf16*)(ws + o_h1), (float*)(ws + o_c1),
      (__bf16*)(ws + o_h2), (float*)(ws + o_c2),
      (float*)(ws + o_Z), (unsigned*)(ws + o_bar));

  // 5) head projection with WMMA
  head_kernel<<<1, 512, 0, stream>>>((const __bf16*)(ws + o_h2),
                                     (const __bf16*)(ws + o_Wdt), bd, out);
}
